// Med2Vec_14826227106204
// MI455X (gfx1250) — matrix operand order repaired and checked by reading, not compiled, not run on hardware
//
#include <hip/hip_runtime.h>
#include <hip/hip_bf16.h>

#define V_   12000
#define D_   200
#define DP   224          // D padded to 7*32 for bf16 WMMA K-steps
#define B_   1024
#define P_   200000
#define EPSF 1e-8f
#define LDST 232          // LDS row stride in bf16 elems (16B aligned, bank-spread)
#define RB   48           // j-rows staged per round (3 WMMA tiles), 250 rounds total
#define JSPL 4            // j-range splits (blockIdx.y) for occupancy

typedef __attribute__((ext_vector_type(16))) __bf16 v16bf;
typedef __attribute__((ext_vector_type(8)))  __bf16 v8bf;
typedef __attribute__((ext_vector_type(8)))  float  v8f;

__device__ __forceinline__ __bf16 f2bf(float f) {
    union { float f; unsigned u; } x; x.f = f;
    unsigned r = x.u + 0x7fffu + ((x.u >> 16) & 1u);   // round-to-nearest-even
    unsigned short h = (unsigned short)(r >> 16);
    __bf16 o; __builtin_memcpy(&o, &h, 2); return o;
}

// two contiguous 16B loads -> one 16-element bf16 fragment register block
__device__ __forceinline__ v16bf ld_frag(const __bf16* p0, const __bf16* p1) {
    v8bf lo = *(const v8bf*)p0;
    v8bf hi = *(const v8bf*)p1;
    return __builtin_shufflevector(lo, hi, 0,1,2,3,4,5,6,7,8,9,10,11,12,13,14,15);
}

// gfx1250 async global->LDS copy (16B), tracked by ASYNCcnt
__device__ __forceinline__ void async_g2l_b128(unsigned lds_off, const void* g) {
    asm volatile("global_load_async_to_lds_b128 %0, %1, off"
                 : : "v"(lds_off), "v"(g) : "memory");
}

// stage one 48x224 bf16 j-block (1344 16B chunks, padded to 1536 so every wave
// issues exactly 6 async instructions -> uniform s_wait_asynccnt immediates)
__device__ __forceinline__ void stage48(const __bf16* __restrict__ pre, int jbase,
                                        unsigned bufbase, unsigned dumpoff, int tid) {
#pragma unroll
    for (int c0 = 0; c0 < 1536; c0 += 256) {
        int c = c0 + tid;
        bool pad = (c >= RB * 28);
        int cc = pad ? 0 : c;
        int row = cc / 28, col = cc - row * 28;            // col in 16B (8-elem) units
        unsigned loff = pad ? dumpoff : (bufbase + (unsigned)((row * LDST + col * 8) * 2));
        async_g2l_b128(loff, pre + (size_t)(jbase + row) * DP + col * 8);
    }
}

// ---------------- init / conversion kernels ----------------
__global__ void k_init(float* acc) {
    if (threadIdx.x == 0 && blockIdx.x == 0) { acc[0] = 0.f; acc[1] = 0.f; }
}
__global__ void k_zero(float* p, int n) {
    int idx = blockIdx.x * blockDim.x + threadIdx.x;
    if (idx < n) p[idx] = 0.f;
}

// pre = bf16(relu(emb)) as [V][DP], zero-padded in d
__global__ void k_cvt_pre(const float* __restrict__ emb, __bf16* __restrict__ pre) {
    int idx = blockIdx.x * blockDim.x + threadIdx.x;
    if (idx >= V_ * DP) return;
    int v = idx / DP, d = idx % DP;
    float val = (d < D_) ? emb[(size_t)v * D_ + d] : 0.f;
    pre[idx] = f2bf(val > 0.f ? val : 0.f);
}

// embT[d][v] = emb[v][d] as [DP][V] (zero rows for d>=200) : B-matrix for GEMM1
__global__ void k_cvt_embT(const float* __restrict__ emb, __bf16* __restrict__ embT) {
    int idx = blockIdx.x * blockDim.x + threadIdx.x;
    if (idx >= DP * V_) return;
    int d = idx / V_, v = idx % V_;
    float val = (d < D_) ? emb[(size_t)v * D_ + d] : 0.f;
    embT[idx] = f2bf(val);
}

// wt[v][d] = W[d][v] as [V][DP], zero-padded in d : B-matrix for GEMM2
__global__ void k_cvt_wt(const float* __restrict__ W, __bf16* __restrict__ wt) {
    int idx = blockIdx.x * blockDim.x + threadIdx.x;
    if (idx >= V_ * DP) return;
    int v = idx / DP, d = idx % DP;
    wt[idx] = f2bf(d < D_ ? W[(size_t)d * V_ + v] : 0.f);
}

__global__ void k_cvt_x(const float* __restrict__ x, __bf16* __restrict__ xb) {
    int idx = blockIdx.x * blockDim.x + threadIdx.x;
    if (idx >= B_ * V_) return;
    xb[idx] = f2bf(x[idx]);
}

// ---------------- GEMM1: vrep = relu(x @ emb), M=1024 K=12000 N=224 ----------------
__global__ void k_gemm1(const __bf16* __restrict__ xb, const __bf16* __restrict__ embT,
                        __bf16* __restrict__ vrep) {
    const int NT = DP / 16;                 // 14
    const int TILES = (B_ / 16) * NT;       // 896
    int wave = (blockIdx.x * blockDim.x + threadIdx.x) >> 5;
    int lane = threadIdx.x & 31;
    if (wave >= TILES) return;
    int m0 = (wave / NT) * 16;
    int n0 = (wave % NT) * 16;
    int half = lane >> 4, l15 = lane & 15;
    const __bf16* arow = xb   + (size_t)(m0 + l15) * V_;
    const __bf16* bcol = embT + (size_t)(n0 + l15) * V_;
    v8f acc = {};
    for (int kb = 0; kb < V_; kb += 32) {
        v16bf a = ld_frag(arow + kb + 8 * half,  arow + kb + 16 + 8 * half);
        v16bf b = ld_frag(bcol + kb + 8 * half, bcol + kb + 16 + 8 * half);
        acc = __builtin_amdgcn_wmma_f32_16x16x32_bf16(false, a, false, b,
                                                      (short)0, acc, false, false);
    }
#pragma unroll
    for (int r = 0; r < 8; ++r) {
        float v = acc[r]; v = v > 0.f ? v : 0.f;        // fused relu
        vrep[(size_t)(m0 + r + 8 * half) * DP + n0 + l15] = f2bf(v);
    }
}

// ------- GEMM2: logits = vrep @ W + b, M=1024 K=224 N=12000, 10 n-tiles/wave -------
__global__ void k_gemm2(const __bf16* __restrict__ vrep, const __bf16* __restrict__ wt,
                        const float* __restrict__ bias, float* __restrict__ logits) {
    const int NG = 75;                      // groups of 10 n-tiles
    const int TILES = (B_ / 16) * NG;       // 4800 waves
    int wave = (blockIdx.x * blockDim.x + threadIdx.x) >> 5;
    int lane = threadIdx.x & 31;
    if (wave >= TILES) return;
    int m0 = (wave / NG) * 16;
    int ng = (wave % NG) * 10;
    int half = lane >> 4, l15 = lane & 15;
    const __bf16* arow = vrep + (size_t)(m0 + l15) * DP;
    v16bf aF[7];
#pragma unroll
    for (int k = 0; k < 7; ++k)
        aF[k] = ld_frag(arow + k * 32 + 8 * half, arow + k * 32 + 16 + 8 * half);
    for (int t = 0; t < 10; ++t) {
        int n0 = (ng + t) * 16;
        const __bf16* bcol = wt + (size_t)(n0 + l15) * DP;
        v8f acc = {};
#pragma unroll
        for (int k = 0; k < 7; ++k) {
            v16bf b = ld_frag(bcol + k * 32 + 8 * half, bcol + k * 32 + 16 + 8 * half);
            acc = __builtin_amdgcn_wmma_f32_16x16x32_bf16(false, aF[k], false, b,
                                                          (short)0, acc, false, false);
        }
        float bv = bias[n0 + l15];
#pragma unroll
        for (int r = 0; r < 8; ++r)
            logits[(size_t)(m0 + r + 8 * half) * V_ + n0 + l15] = acc[r] + bv;
    }
}

// ---------------- softmax rows (in place), one block per row ----------------
__global__ void k_softmax(float* __restrict__ logits) {
    __shared__ float red[256];
    int row = blockIdx.x, tid = threadIdx.x;
    float* p = logits + (size_t)row * V_;
    float mx = -3.0e38f;
    for (int i = tid; i < V_; i += 256) mx = fmaxf(mx, p[i]);
    red[tid] = mx; __syncthreads();
    for (int s = 128; s > 0; s >>= 1) { if (tid < s) red[tid] = fmaxf(red[tid], red[tid + s]); __syncthreads(); }
    mx = red[0]; __syncthreads();
    float sum = 0.f;
    for (int i = tid; i < V_; i += 256) { float e = __expf(p[i] - mx); p[i] = e; sum += e; }
    red[tid] = sum; __syncthreads();
    for (int s = 128; s > 0; s >>= 1) { if (tid < s) red[tid] += red[tid + s]; __syncthreads(); }
    float inv = 1.f / red[0];
    for (int i = tid; i < V_; i += 256) p[i] *= inv;
}

// ---------------- visit cross-entropy sum -> acc[1] ----------------
__global__ void k_visitcost(const float* __restrict__ x, const float* __restrict__ mask,
                            const float* __restrict__ pred, float* __restrict__ acc) {
    const long long N = (long long)(B_ - 1) * V_;
    long long stride = (long long)gridDim.x * blockDim.x;
    float s = 0.f;
    for (long long n = (long long)blockIdx.x * blockDim.x + threadIdx.x; n < N; n += stride) {
        int t = (int)(n / V_), v = (int)(n % V_);
        float m1  = mask[t] * mask[t + 1];
        float fwd = pred[(size_t)t * V_ + v] * m1;
        float xf  = x[(size_t)(t + 1) * V_ + v];
        s -= xf * __logf(fwd + EPSF) + (1.f - xf) * __logf(1.f - fwd + EPSF);
        float bwd = pred[(size_t)(t + 1) * V_ + v] * m1;
        float xb  = x[(size_t)t * V_ + v];
        s -= xb * __logf(bwd + EPSF) + (1.f - xb) * __logf(1.f - bwd + EPSF);
    }
    for (int o = 16; o > 0; o >>= 1) s += __shfl_xor(s, o, 32);
    if ((threadIdx.x & 31) == 0) atomicAdd(&acc[1], s);
}

// ---- norms[i] += sum_{j in split} exp(pre_i . pre_j)
//      8 i-tiles/block, j split over blockIdx.y, double-buffered async-LDS B ----
__global__ void __launch_bounds__(256) k_norms(const __bf16* __restrict__ pre,
                                               float* __restrict__ norms) {
    __shared__ __align__(16) __bf16 bt[2 * RB * LDST];   // two 48x224 j-blocks
    __shared__ __align__(16) __bf16 dump[8];             // sink for padded DMA chunks
    int wave = threadIdx.x >> 5;                         // 0..7
    int lane = threadIdx.x & 31;
    int tile = blockIdx.x * 8 + wave;
    bool active = tile < (V_ / 16);
    int i0 = (active ? tile : (V_ / 16) - 1) * 16;       // clamp: uniform barriers/EXEC
    int half = lane >> 4, l15 = lane & 15;
    const __bf16* arow = pre + (size_t)(i0 + l15) * DP;
    v16bf aF[7];
#pragma unroll
    for (int k = 0; k < 7; ++k)
        aF[k] = ld_frag(arow + k * 32 + 8 * half, arow + k * 32 + 16 + 8 * half);

    unsigned base0   = (unsigned)(uintptr_t)&bt[0];
    unsigned base1   = (unsigned)(uintptr_t)&bt[RB * LDST];
    unsigned dumpoff = (unsigned)(uintptr_t)&dump[0];

    const int NROUND = V_ / RB;                          // 250 rounds of 48 rows
    int r0 = ((int)blockIdx.y * NROUND) / JSPL;          // this block's j-range
    int r1 = (((int)blockIdx.y + 1) * NROUND) / JSPL;

    float es[8] = {0.f, 0.f, 0.f, 0.f, 0.f, 0.f, 0.f, 0.f};
    stage48(pre, r0 * RB, base0, dumpoff, threadIdx.x);  // prologue: block r0 -> buf0
    for (int r = r0; r < r1; ++r) {
        if (r + 1 < r1) {                                // prefetch next into other buf
            stage48(pre, (r + 1) * RB, ((r - r0) & 1) ? base0 : base1, dumpoff, threadIdx.x);
            asm volatile("s_wait_asynccnt 6" ::: "memory");   // current buf landed
        } else {
            asm volatile("s_wait_asynccnt 0" ::: "memory");
        }
        __syncthreads();
        const __bf16* buf = bt + (((r - r0) & 1) ? RB * LDST : 0);
#pragma unroll
        for (int t = 0; t < 3; ++t) {
            const __bf16* bcol = buf + (t * 16 + l15) * LDST;
            v8f acc = {};
#pragma unroll
            for (int k = 0; k < 7; ++k) {
                v16bf b = ld_frag(bcol + k * 32 + 8 * half, bcol + k * 32 + 16 + 8 * half);
                acc = __builtin_amdgcn_wmma_f32_16x16x32_bf16(false, aF[k], false, b,
                                                              (short)0, acc, false, false);
            }
#pragma unroll
            for (int q = 0; q < 8; ++q) es[q] += __expf(acc[q]);
        }
        __syncthreads();                                 // done reading buf before refill
    }
#pragma unroll
    for (int q = 0; q < 8; ++q) {                        // reduce over 16 N-lanes per half
        float s = es[q];
        for (int o = 1; o < 16; o <<= 1) s += __shfl_xor(s, o, 32);
        if (active && l15 == 0) atomicAdd(&norms[i0 + 8 * half + q], s);
    }
}

// ---------------- pair gather cost -> acc[0] ----------------
__global__ void k_embcost(const float* __restrict__ emb, const int* __restrict__ iv,
                          const int* __restrict__ jv, const float* __restrict__ norms,
                          float* __restrict__ acc) {
    int gwave  = (blockIdx.x * blockDim.x + threadIdx.x) >> 5;
    int lane   = threadIdx.x & 31;
    int nwaves = (gridDim.x * blockDim.x) >> 5;
    float s = 0.f;
    for (int p = gwave; p < P_; p += nwaves) {
        int i = iv[p], j = jv[p];
        float dot = 0.f;
        for (int d = lane; d < D_; d += 32) {
            float a = emb[(size_t)i * D_ + d]; a = a > 0.f ? a : 0.f;
            float b = emb[(size_t)j * D_ + d]; b = b > 0.f ? b : 0.f;
            dot += a * b;
        }
        for (int o = 16; o > 0; o >>= 1) dot += __shfl_xor(dot, o, 32);
        if (lane == 0) s -= __logf(__expf(dot) / norms[i] + EPSF);
    }
    if (lane == 0) atomicAdd(&acc[0], s);
}

// ---------------- final scalar ----------------
__global__ void k_final(const float* __restrict__ mask, const float* __restrict__ acc,
                        float* __restrict__ out) {
    if (threadIdx.x == 0 && blockIdx.x == 0) {
        float msum = 0.f;
        for (int t = 0; t < B_ - 1; ++t) msum += mask[t] * mask[t + 1];
        out[0] = acc[0] / (float)P_ + acc[1] / (msum + EPSF);
    }
}

static inline size_t align256(size_t v) { return (v + 255) & ~(size_t)255; }

extern "C" void kernel_launch(void* const* d_in, const int* in_sizes, int n_in,
                              void* d_out, int out_size, void* d_ws, size_t ws_size,
                              hipStream_t stream) {
    (void)in_sizes; (void)n_in; (void)out_size; (void)ws_size;
    const float* x    = (const float*)d_in[0];
    const float* mask = (const float*)d_in[1];
    const float* emb  = (const float*)d_in[2];
    const float* W    = (const float*)d_in[3];
    const float* bias = (const float*)d_in[4];
    const int*   iv   = (const int*)d_in[5];
    const int*   jv   = (const int*)d_in[6];
    float* out = (float*)d_out;

    char* ws = (char*)d_ws;
    size_t o = 0;
    __bf16* pre    = (__bf16*)(ws + o); o = align256(o + (size_t)V_ * DP * 2);
    __bf16* embT   = (__bf16*)(ws + o); o = align256(o + (size_t)DP * V_ * 2);
    __bf16* wt     = (__bf16*)(ws + o); o = align256(o + (size_t)V_ * DP * 2);
    __bf16* xb     = (__bf16*)(ws + o); o = align256(o + (size_t)B_ * V_ * 2);
    __bf16* vrep   = (__bf16*)(ws + o); o = align256(o + (size_t)B_ * DP * 2);
    float*  logits = (float*)(ws + o);  o = align256(o + (size_t)B_ * V_ * 4);
    float*  norms  = (float*)(ws + o);  o = align256(o + (size_t)V_ * 4);
    float*  acc    = (float*)(ws + o);  o = align256(o + 2 * 4);

    k_init<<<1, 1, 0, stream>>>(acc);
    k_zero<<<(V_ + 255) / 256, 256, 0, stream>>>(norms, V_);
    k_cvt_pre <<<(V_ * DP + 255) / 256, 256, 0, stream>>>(emb, pre);
    k_cvt_embT<<<(DP * V_ + 255) / 256, 256, 0, stream>>>(emb, embT);
    k_cvt_wt  <<<(V_ * DP + 255) / 256, 256, 0, stream>>>(W, wt);
    k_cvt_x   <<<(B_ * V_ + 255) / 256, 256, 0, stream>>>(x, xb);

    k_gemm1<<<(896 + 3) / 4, 128, 0, stream>>>(xb, embT, vrep);        // 4 waves/block
    k_gemm2<<<4800 / 4, 128, 0, stream>>>(vrep, wt, bias, logits);
    k_softmax<<<B_, 256, 0, stream>>>(logits);
    k_visitcost<<<1024, 256, 0, stream>>>(x, mask, logits, acc);

    dim3 ngrid((750 + 7) / 8, JSPL);                    // 94 i-blocks x 4 j-splits
    k_norms<<<ngrid, 256, 0, stream>>>(pre, norms);
    k_embcost<<<512, 256, 0, stream>>>(emb, iv, jv, norms, acc);

    k_final<<<1, 1, 0, stream>>>(mask, acc, out);
}
